// MoEV3_34935263986344
// MI455X (gfx1250) — matrix operand order repaired and checked by reading, not compiled, not run on hardware
//
#include <hip/hip_runtime.h>
#include <hip/hip_bf16.h>

// ---------------------------------------------------------------------------
// MoE (DeepSeek-V3 style) forward for gfx1250 (MI455X), wave32 + WMMA bf16,
// software-pipelined staging + async global->LDS (ASYNCcnt) for bf16 tiles.
// ---------------------------------------------------------------------------

typedef __bf16 bf16_t;
typedef __attribute__((ext_vector_type(16))) __bf16 v16bf;
typedef __attribute__((ext_vector_type(8)))  __bf16 v8bf;
typedef __attribute__((ext_vector_type(8)))  float  v8f;

constexpr int DIM   = 1024;
constexpr int INTER = 1024;
constexpr int NE    = 32;     // routed experts
constexpr int TOPK  = 4;
constexpr int NTOK  = 8192;   // B*S
constexpr int CAP   = 1280;   // int(1.25 * NTOK*TOPK / NE)
static_assert(CAP == (NTOK * TOPK * 5) / (NE * 4), "capacity formula");

constexpr int BM = 128, BN = 64, BK = 32;
constexpr int LDA = 40;  // As row stride (bf16 elems), 80B -> 16B aligned, padded
constexpr int LDB = 40;  // Bs (transposed) row stride

__device__ __forceinline__ v16bf cat16(v8bf lo, v8bf hi) {
  return __builtin_shufflevector(lo, hi, 0,1,2,3,4,5,6,7,8,9,10,11,12,13,14,15);
}

// Truncate a generic pointer to a __shared__ object to its LDS byte offset
// (flat LDS aperture keeps the allocation offset in addr[31:0]).
__device__ __forceinline__ unsigned lds_addr(const void* p) {
  return (unsigned)(uintptr_t)p;
}

// 32B/lane async DMA: global -> LDS, tracked with ASYNCcnt.
__device__ __forceinline__ void async_copy_32B(unsigned lds, const void* gptr) {
  asm volatile(
      "global_load_async_to_lds_b128 %0, %1, off\n\t"
      "global_load_async_to_lds_b128 %0, %1, off offset:16"
      :: "v"(lds), "v"(gptr) : "memory");
}
__device__ __forceinline__ void wait_async_le2() {
  asm volatile("s_wait_asynccnt 0x2" ::: "memory");
}
__device__ __forceinline__ void wait_async_0() {
  asm volatile("s_wait_asynccnt 0x0" ::: "memory");
}

// ---------------------------------------------------------------------------
// 1) Gating: one wave per token, lane = expert.
// ---------------------------------------------------------------------------
__global__ __launch_bounds__(256) void gate_kernel(
    const float* __restrict__ X, const float* __restrict__ WG,
    int* __restrict__ idxOut, float* __restrict__ wOut)
{
  const int lane = threadIdx.x & 31;
  const int wid  = threadIdx.x >> 5;
  const int t    = blockIdx.x * 8 + wid;

  const float4* xr = (const float4*)(X  + (size_t)t    * DIM);
  const float4* gr = (const float4*)(WG + (size_t)lane * DIM);
  float logit = 0.f;
  #pragma unroll 4
  for (int d = 0; d < DIM / 4; ++d) {
    float4 a = xr[d], b = gr[d];
    logit += a.x*b.x + a.y*b.y + a.z*b.z + a.w*b.w;
  }

  // softmax across 32 lanes
  float m = logit;
  #pragma unroll
  for (int off = 16; off > 0; off >>= 1) m = fmaxf(m, __shfl_xor(m, off, 32));
  float p = __expf(logit - m);
  float s = p;
  #pragma unroll
  for (int off = 16; off > 0; off >>= 1) s += __shfl_xor(s, off, 32);
  const float score = p / s;

  // group max over aligned quads (E/G = 4)
  float gm = score;
  gm = fmaxf(gm, __shfl_xor(gm, 1, 32));
  gm = fmaxf(gm, __shfl_xor(gm, 2, 32));

  // top-4 of 8 group scores (lowest-index ties), computed redundantly per lane
  float gval[8];
  #pragma unroll
  for (int g = 0; g < 8; ++g) gval[g] = __shfl(gm, g * 4, 32);
  unsigned keep = 0;
  #pragma unroll
  for (int k = 0; k < 4; ++k) {
    float best = -1e30f; int bi = 0;
    #pragma unroll
    for (int g = 0; g < 8; ++g)
      if (!((keep >> g) & 1) && gval[g] > best) { best = gval[g]; bi = g; }
    keep |= 1u << bi;
  }

  float masked = ((keep >> (lane >> 2)) & 1) ? score : -1e30f;

  // top-4 experts, lowest-index ties; weights gathered from ORIGINAL scores
  float selW[4]; int selI[4];
  float v = masked;
  #pragma unroll
  for (int k = 0; k < 4; ++k) {
    float bv = v; int bidx = lane;
    #pragma unroll
    for (int off = 16; off > 0; off >>= 1) {
      float ov = __shfl_xor(bv, off, 32);
      int   oi = __shfl_xor(bidx, off, 32);
      if (ov > bv || (ov == bv && oi < bidx)) { bv = ov; bidx = oi; }
    }
    selI[k] = bidx;
    selW[k] = __shfl(score, bidx, 32);
    if (lane == bidx) v = -1e30f;
  }
  const float inv = 1.f / (selW[0] + selW[1] + selW[2] + selW[3] + 1e-9f); // ROUTE_SCALE=1
  #pragma unroll
  for (int k = 0; k < 4; ++k) {
    if (lane == k) {
      idxOut[t * TOPK + k] = selI[k];
      wOut  [t * TOPK + k] = selW[k] * inv;
    }
  }
}

// ---------------------------------------------------------------------------
// 2) Dispatch: 32 waves, wave = expert; token-major rank scan with capacity.
// ---------------------------------------------------------------------------
__global__ __launch_bounds__(1024) void dispatch_kernel(
    const int* __restrict__ idxF, const float* __restrict__ wF,
    int* __restrict__ bufTok, float* __restrict__ bufW)
{
  const int lane = threadIdx.x & 31;
  const int e    = threadIdx.x >> 5;

  for (int s = lane; s < CAP; s += 32) {  // prefill pad slots
    bufTok[e * CAP + s] = NTOK;
    bufW  [e * CAP + s] = 0.f;
  }
  __syncthreads();

  int c = 0;
  const unsigned lt = (1u << lane) - 1u;
  for (int base = 0; base < NTOK * TOPK; base += 32) {
    const int i  = base + lane;
    const int fe = idxF[i];
    const bool match = (fe == e);
    const unsigned mask = (unsigned)__ballot(match);
    const int pos = c + __popc(mask & lt);
    if (match && pos < CAP) {
      bufTok[e * CAP + pos] = i >> 2;  // token = i / TOPK
      bufW  [e * CAP + pos] = wF[i];
    }
    c += __popc(mask);
  }
}

// ---------------------------------------------------------------------------
// 3) Grouped GEMM1: H = silu(X@W1) * (X@W3), dual WMMA accumulators,
//    register-pipelined staging (next tile's global loads overlap WMMA).
// ---------------------------------------------------------------------------
template<bool GATHER>
__global__ __launch_bounds__(256) void ffn_gemm1(
    const float* __restrict__ X,
    const float* __restrict__ W1, const float* __restrict__ W3,
    const int* __restrict__ bufTok,
    bf16_t* __restrict__ H, int rowsPerE)
{
  __shared__ bf16_t As [BM * LDA];
  __shared__ bf16_t Bs1[BN * LDB];
  __shared__ bf16_t Bs3[BN * LDB];

  const int tid     = threadIdx.x;
  const int e       = blockIdx.z;
  const int rowBase = e * rowsPerE + blockIdx.y * BM;
  const int colBase = blockIdx.x * BN;
  const float* W1e = W1 + (size_t)e * DIM * INTER;
  const float* W3e = W3 + (size_t)e * DIM * INTER;

  const int arow = tid >> 1;
  const int akc  = (tid & 1) * 16;
  int tok = GATHER ? bufTok[rowBase + arow] : (rowBase + arow);
  const bool tokOk = tok < NTOK;
  const float* xrow = X + (size_t)tok * DIM;

  const int bk = tid >> 3;
  const int bn = (tid & 7) * 8;

  const int lane = tid & 31;
  const int wM   = (tid >> 5) & 3;
  const int wN   = (tid >> 5) >> 2;
  const int lh   = lane & 15;
  const int kbA  = (lane < 16) ? 0 : 8;
  const int kbB  = (lane < 16) ? 0 : 8;

  v8f acc1[2][2] = {}, acc3[2][2] = {};

  float4 aR[4];            // A staging regs (16 f32)
  float4 b1R[2], b3R[2];   // B staging regs (8 f32 each)

  auto loadA = [&](int k0) {
    if (tokOk) {
      const float4* p = (const float4*)(xrow + k0 + akc);
      #pragma unroll
      for (int j = 0; j < 4; ++j) aR[j] = p[j];
    } else {
      #pragma unroll
      for (int j = 0; j < 4; ++j) aR[j] = make_float4(0.f, 0.f, 0.f, 0.f);
    }
  };
  auto loadB = [&](int k0) {
    const float4* p1 = (const float4*)(W1e + (size_t)(k0 + bk) * INTER + colBase + bn);
    const float4* p3 = (const float4*)(W3e + (size_t)(k0 + bk) * INTER + colBase + bn);
    b1R[0] = p1[0]; b1R[1] = p1[1];
    b3R[0] = p3[0]; b3R[1] = p3[1];
  };
  auto stageA = [&]() {
    bf16_t* dst = &As[arow * LDA + akc];
    #pragma unroll
    for (int j = 0; j < 4; ++j) {
      dst[j*4+0] = (bf16_t)aR[j].x; dst[j*4+1] = (bf16_t)aR[j].y;
      dst[j*4+2] = (bf16_t)aR[j].z; dst[j*4+3] = (bf16_t)aR[j].w;
    }
  };
  auto stageB = [&]() {
    float f1[8] = {b1R[0].x,b1R[0].y,b1R[0].z,b1R[0].w,b1R[1].x,b1R[1].y,b1R[1].z,b1R[1].w};
    float f3[8] = {b3R[0].x,b3R[0].y,b3R[0].z,b3R[0].w,b3R[1].x,b3R[1].y,b3R[1].z,b3R[1].w};
    #pragma unroll
    for (int j = 0; j < 8; ++j) {
      Bs1[(bn + j) * LDB + bk] = (bf16_t)f1[j];
      Bs3[(bn + j) * LDB + bk] = (bf16_t)f3[j];
    }
  };

  constexpr int NIT = DIM / BK;
  loadA(0); loadB(0);

  for (int it = 0; it < NIT; ++it) {
    stageA(); stageB();
    __syncthreads();
    if (it + 1 < NIT) { loadA((it + 1) * BK); loadB((it + 1) * BK); }

    v16bf af[2], b1f[2], b3f[2];
    #pragma unroll
    for (int mt = 0; mt < 2; ++mt) {
      const bf16_t* pa = &As[(wM * 32 + mt * 16 + lh) * LDA];
      af[mt] = cat16(*(const v8bf*)(pa + kbA), *(const v8bf*)(pa + 16 + kbA));
    }
    #pragma unroll
    for (int nt = 0; nt < 2; ++nt) {
      const bf16_t* p1 = &Bs1[(wN * 32 + nt * 16 + lh) * LDB + kbB];
      const bf16_t* p3 = &Bs3[(wN * 32 + nt * 16 + lh) * LDB + kbB];
      b1f[nt] = cat16(*(const v8bf*)p1, *(const v8bf*)(p1 + 16));
      b3f[nt] = cat16(*(const v8bf*)p3, *(const v8bf*)(p3 + 16));
    }
    #pragma unroll
    for (int mt = 0; mt < 2; ++mt)
      #pragma unroll
      for (int nt = 0; nt < 2; ++nt) {
        acc1[mt][nt] = __builtin_amdgcn_wmma_f32_16x16x32_bf16(
            false, af[mt], false, b1f[nt], (short)0, acc1[mt][nt], false, false);
        acc3[mt][nt] = __builtin_amdgcn_wmma_f32_16x16x32_bf16(
            false, af[mt], false, b3f[nt], (short)0, acc3[mt][nt], false, false);
      }
    __syncthreads();
  }

  const int mofs = (lane < 16) ? 0 : 8;
  #pragma unroll
  for (int mt = 0; mt < 2; ++mt)
    #pragma unroll
    for (int nt = 0; nt < 2; ++nt)
      #pragma unroll
      for (int i = 0; i < 8; ++i) {
        const int r = rowBase + wM * 32 + mt * 16 + i + mofs;
        const int c = colBase + wN * 32 + nt * 16 + lh;
        const float g = acc1[mt][nt][i];
        const float u = acc3[mt][nt][i];
        H[(size_t)r * INTER + c] = (bf16_t)((g / (1.f + __expf(-g))) * u);
      }
}

// ---------------------------------------------------------------------------
// 4) Grouped GEMM2: OUT = H@W2. A-tile (bf16) streamed with
//    GLOBAL_LOAD_ASYNC_TO_LDS_B128 into a double buffer; B register-pipelined.
//    Routed path scales by buf_w and atomic-scatters into OUT.
// ---------------------------------------------------------------------------
template<bool SCATTER>
__global__ __launch_bounds__(256) void ffn_gemm2(
    const bf16_t* __restrict__ H, const float* __restrict__ W2,
    const int* __restrict__ bufTok, const float* __restrict__ bufW,
    float* __restrict__ OUT, int rowsPerE)
{
  __shared__ bf16_t As[2][BM * LDA];
  __shared__ bf16_t Bs[BN * LDB];

  const int tid     = threadIdx.x;
  const int e       = blockIdx.z;
  const int rowBase = e * rowsPerE + blockIdx.y * BM;
  const int colBase = blockIdx.x * BN;
  const float* W2e = W2 + (size_t)e * INTER * DIM;

  const int arow = tid >> 1;
  const int akc  = (tid & 1) * 16;
  const bf16_t* hrow = H + (size_t)(rowBase + arow) * INTER;
  const unsigned ldsA0 = lds_addr(&As[0][arow * LDA + akc]);
  const unsigned ldsA1 = lds_addr(&As[1][arow * LDA + akc]);

  const int bk = tid >> 3;
  const int bn = (tid & 7) * 8;

  const int lane = tid & 31;
  const int wM   = (tid >> 5) & 3;
  const int wN   = (tid >> 5) >> 2;
  const int lh   = lane & 15;
  const int kbA  = (lane < 16) ? 0 : 8;
  const int kbB  = (lane < 16) ? 0 : 8;

  v8f acc[2][2] = {};
  float4 bR[2];

  auto loadB = [&](int k0) {
    const float4* p = (const float4*)(W2e + (size_t)(k0 + bk) * DIM + colBase + bn);
    bR[0] = p[0]; bR[1] = p[1];
  };
  auto stageB = [&]() {
    float f[8] = {bR[0].x,bR[0].y,bR[0].z,bR[0].w,bR[1].x,bR[1].y,bR[1].z,bR[1].w};
    #pragma unroll
    for (int j = 0; j < 8; ++j) Bs[(bn + j) * LDB + bk] = (bf16_t)f[j];
  };

  constexpr int NIT = INTER / BK;
  async_copy_32B(ldsA0, hrow + akc);   // A tile 0 -> As[0] (ASYNCcnt += 2)
  loadB(0);

  for (int it = 0; it < NIT; ++it) {
    const int cur = it & 1;
    stageB();
    if (it + 1 < NIT) {
      // DMA next A tile into the other buffer; overlaps this tile's WMMAs.
      async_copy_32B(cur ? ldsA0 : ldsA1, hrow + (it + 1) * BK + akc);
      wait_async_le2();   // in-order: current tile's pair has landed
    } else {
      wait_async_0();
    }
    __syncthreads();
    if (it + 1 < NIT) loadB((it + 1) * BK);

    const bf16_t* Ac = As[cur];
    v16bf af[2], bf[2];
    #pragma unroll
    for (int mt = 0; mt < 2; ++mt) {
      const bf16_t* pa = &Ac[(wM * 32 + mt * 16 + lh) * LDA];
      af[mt] = cat16(*(const v8bf*)(pa + kbA), *(const v8bf*)(pa + 16 + kbA));
    }
    #pragma unroll
    for (int nt = 0; nt < 2; ++nt) {
      const bf16_t* pb = &Bs[(wN * 32 + nt * 16 + lh) * LDB + kbB];
      bf[nt] = cat16(*(const v8bf*)pb, *(const v8bf*)(pb + 16));
    }
    #pragma unroll
    for (int mt = 0; mt < 2; ++mt)
      #pragma unroll
      for (int nt = 0; nt < 2; ++nt)
        acc[mt][nt] = __builtin_amdgcn_wmma_f32_16x16x32_bf16(
            false, af[mt], false, bf[nt], (short)0, acc[mt][nt], false, false);
    __syncthreads();
  }

  const int mofs = (lane < 16) ? 0 : 8;
  #pragma unroll
  for (int mt = 0; mt < 2; ++mt)
    #pragma unroll
    for (int nt = 0; nt < 2; ++nt)
      #pragma unroll
      for (int i = 0; i < 8; ++i) {
        const int r = rowBase + wM * 32 + mt * 16 + i + mofs;
        const int c = colBase + wN * 32 + nt * 16 + lh;
        const float v = acc[mt][nt][i];
        if (SCATTER) {
          const int tok = bufTok[r];
          if (tok < NTOK)
            atomicAdd(&OUT[(size_t)tok * DIM + c], v * bufW[r]);
        } else {
          OUT[(size_t)r * DIM + c] = v;
        }
      }
}

// ---------------------------------------------------------------------------
// Launch: inputs = x, wg, w1, w2, w3, sw1, sw2, sw3 (setup_inputs order).
// ---------------------------------------------------------------------------
extern "C" void kernel_launch(void* const* d_in, const int* in_sizes, int n_in,
                              void* d_out, int out_size, void* d_ws, size_t ws_size,
                              hipStream_t stream) {
  const float* x   = (const float*)d_in[0];
  const float* wg  = (const float*)d_in[1];
  const float* w1  = (const float*)d_in[2];
  const float* w2  = (const float*)d_in[3];
  const float* w3  = (const float*)d_in[4];
  const float* sw1 = (const float*)d_in[5];
  const float* sw2 = (const float*)d_in[6];
  const float* sw3 = (const float*)d_in[7];
  float* out = (float*)d_out;
  (void)in_sizes; (void)n_in; (void)out_size; (void)ws_size;

  char* ws = (char*)d_ws;
  size_t off = 0;
  auto carve = [&](size_t bytes) -> void* {
    off = (off + 255) & ~(size_t)255;
    void* p = ws + off;
    off += bytes;
    return p;
  };
  int*    idxF   = (int*)   carve((size_t)NTOK * TOPK * sizeof(int));
  float*  wF     = (float*) carve((size_t)NTOK * TOPK * sizeof(float));
  int*    bufTok = (int*)   carve((size_t)NE * CAP * sizeof(int));
  float*  bufW   = (float*) carve((size_t)NE * CAP * sizeof(float));
  bf16_t* Hs     = (bf16_t*)carve((size_t)NTOK * INTER * sizeof(bf16_t));     // 16 MB
  bf16_t* He     = (bf16_t*)carve((size_t)NE * CAP * INTER * sizeof(bf16_t)); // 80 MB

  // 1) router
  gate_kernel<<<NTOK / 8, 256, 0, stream>>>(x, wg, idxF, wF);
  // 2) capacity dispatch
  dispatch_kernel<<<1, 1024, 0, stream>>>(idxF, wF, bufTok, bufW);
  // 3) shared expert: writes d_out (plain stores) first
  ffn_gemm1<false><<<dim3(INTER / BN, NTOK / BM, 1), 256, 0, stream>>>(
      x, sw1, sw3, nullptr, Hs, NTOK);
  ffn_gemm2<false><<<dim3(DIM / BN, NTOK / BM, 1), 256, 0, stream>>>(
      Hs, sw2, nullptr, nullptr, out, NTOK);
  // 4) routed experts: atomic scatter-add on top
  ffn_gemm1<true><<<dim3(INTER / BN, CAP / BM, NE), 256, 0, stream>>>(
      x, w1, w3, bufTok, He, CAP);
  ffn_gemm2<true><<<dim3(DIM / BN, CAP / BM, NE), 256, 0, stream>>>(
      He, w2, bufTok, bufW, out, CAP);
}
